// NGFKT_74758200754434
// MI455X (gfx1250) — matrix operand order repaired and checked by reading, not compiled, not run on hardware
//
#include <hip/hip_runtime.h>
#include <hip/hip_bf16.h>
#include <math.h>

// ---------------------------------------------------------------------------
// NGFKT forward for MI455X (gfx1250, wave32).
// Shapes: B=16, S=512, E=256, H=8, D=32, L=2.
// d_out = [final (8192 f32)] ++ [attn (16*8*512*512 f32)]
// ---------------------------------------------------------------------------

#define BB 16
#define SS 512
#define EE 256
#define HH 8
#define DD 32
#define NEGV (-1e9f)

typedef __attribute__((ext_vector_type(16))) _Float16 v16h;
typedef __attribute__((ext_vector_type(2)))  _Float16 v2h;
typedef __attribute__((ext_vector_type(8)))  float    v8f;

__device__ __forceinline__ v8f wmma_f16(v16h a, v16h b, v8f c) {
  // D(f32 16x16) = A(f16 16x32) * B(f16 32x16) + C
  return __builtin_amdgcn_wmma_f32_16x16x32_f16(
      /*neg_a=*/false, a, /*neg_b=*/false, b,
      /*c_mod=*/(short)0, c, /*reuse_a=*/false, /*reuse_b=*/false);
}

// A fragment: 16x32, element [m][k] at s[m*ld + k]  (f32 -> f16 convert)
// Layout (ISA 7.12.2): lanes 0-15 row m hold K=0..7,16..23; lanes 16-31 K=8..15,24..31
__device__ __forceinline__ v16h load_a_frag(const float* s, int ld) {
  int lane = threadIdx.x & 31;
  int m  = lane & 15;
  int kh = (lane >> 4) * 8;
  v16h a;
#pragma unroll
  for (int i = 0; i < 8; ++i) {
    a[i]     = (_Float16)s[m * ld + kh + i];
    a[i + 8] = (_Float16)s[m * ld + 16 + kh + i];
  }
  return a;
}

// B fragment (row-major K x N source): B[k][n] = s[k*ld + coloff + n], K=32,N=16
__device__ __forceinline__ v16h load_b_frag(const float* s, int ld, int coloff) {
  int lane = threadIdx.x & 31;
  int n  = lane & 15;
  int kb = (lane >> 4) * 8;
  v16h b;
#pragma unroll
  for (int j = 0; j < 16; ++j)
    b[j] = (_Float16)s[(kb + (j & 7) + ((j >> 3) << 4)) * ld + coloff + n];
  return b;
}

// B fragment from transposed source (key tile): B[k][n] = s[n*ld + k]
__device__ __forceinline__ v16h load_b_frag_T(const float* s, int ld) {
  int lane = threadIdx.x & 31;
  int n  = lane & 15;
  int kb = (lane >> 4) * 8;
  v16h b;
#pragma unroll
  for (int j = 0; j < 16; ++j)
    b[j] = (_Float16)s[n * ld + kb + (j & 7) + ((j >> 3) << 4)];
  return b;
}

// Async global->LDS staging of a rows x cols f32 tile (one wave), CDNA5 path.
__device__ __forceinline__ void async_stage_f32(const float* gsrc, int rows, int cols,
                                                int gld, float* sdst, int sld) {
  int lane  = threadIdx.x & 31;
  int total = rows * cols;
  for (int t = lane; t < total; t += 32) {
    int r = t / cols;
    int c = t - r * cols;
    unsigned lds = (unsigned)(size_t)(sdst + r * sld + c);           // low 32 bits = LDS offset
    unsigned long long ga = (unsigned long long)(size_t)(gsrc + (size_t)r * gld + c);
    asm volatile("global_load_async_to_lds_b32 %0, %1, off"
                 :: "v"(lds), "v"(ga) : "memory");
  }
  asm volatile("s_wait_asynccnt 0x0" ::: "memory");
}

// ---------------------------------------------------------------------------
// 0) gather: inputs = [emb*lab , emb*(1-lab)], query = emb[item_ids]
// ---------------------------------------------------------------------------
__global__ __launch_bounds__(256)
void ngfkt_gather_kernel(const float* __restrict__ emb,
                         const int* __restrict__ item_in,
                         const int* __restrict__ lab_in,
                         const int* __restrict__ ids,
                         float* __restrict__ inputs, float* __restrict__ query) {
  int idx = blockIdx.x * blockDim.x + threadIdx.x;  // over B*S*E
  if (idx >= BB * SS * EE) return;
  int bs = idx >> 8;
  int e  = idx & 255;
  float ie  = emb[(size_t)item_in[bs] * EE + e];
  float lab = (float)lab_in[bs];
  inputs[(size_t)bs * (2 * EE) + e]        = ie * lab;
  inputs[(size_t)bs * (2 * EE) + EE + e]   = ie * (1.f - lab);
  query[(size_t)bs * EE + e] = emb[(size_t)ids[bs] * EE + e];
}

// ---------------------------------------------------------------------------
// 1) WMMA GEMM: C[M,N] = act(A[M,K] @ W[K,N] + bias), act: 0 none,1 relu,2 sigm.
//    64x64 tile per WG (4 waves), K step 32.
//    LDS tiles are stored as f16 ALREADY IN FRAGMENT ORDER [lane][16 halves],
//    so the inner loop is pure ds_load_b128 -> v_wmma (no per-use converts).
// ---------------------------------------------------------------------------
__global__ __launch_bounds__(128)
void ngfkt_gemm_kernel(const float* __restrict__ A, const float* __restrict__ W,
                       const float* __restrict__ bias, float* __restrict__ C,
                       int M, int N, int K, int act) {
  // [row-group 0..3][lane][half]  /  [n-tile 0..3][lane][half]
  __shared__ __align__(32) _Float16 sAh[4][32][16];
  __shared__ __align__(32) _Float16 sWh[4][32][16];
  int wave = threadIdx.x >> 5;
  int lane = threadIdx.x & 31;
  int m0 = blockIdx.x * 64;
  int n0 = blockIdx.y * 64;

  v8f acc[4] = {};

  for (int kc = 0; kc < K; kc += 32) {
    // ---- stage A chunk 64x32 -> f16 fragment layout (float4 global reads)
    for (int t = threadIdx.x; t < 512; t += 128) {
      int r  = t >> 3;            // 0..63
      int c4 = (t & 7) * 4;       // 0,4,..,28
      float4 v = *(const float4*)(A + (size_t)(m0 + r) * K + kc + c4);
      int g  = r >> 4;
      int la = (r & 15) + (((c4 >> 3) & 1) << 4);   // +16 lanes for K 8..15 / 24..31
      int i0 = (c4 & 7) + ((c4 >> 4) << 3);         // 0..7 for K<16, 8..15 for K>=16
      v2h p0; p0[0] = (_Float16)v.x; p0[1] = (_Float16)v.y;
      v2h p1; p1[0] = (_Float16)v.z; p1[1] = (_Float16)v.w;
      *(v2h*)&sAh[g][la][i0]     = p0;
      *(v2h*)&sAh[g][la][i0 + 2] = p1;
    }
    // ---- stage W chunk 32x64 -> f16 fragment layout
    for (int t = threadIdx.x; t < 512; t += 128) {
      int r  = t >> 4;            // 0..31 (K)
      int c4 = (t & 15) * 4;      // 0..60 (N)
      float4 v = *(const float4*)(W + (size_t)(kc + r) * N + n0 + c4);
      int idx  = (r & 7) + ((r >> 4) << 3);
      int hi   = ((r >> 3) & 1) << 4;   // +16 lanes for K rows 8..15 / 24..31
      int tile = c4 >> 4;
      int lb   = (c4 & 15) + hi;
      sWh[tile][lb + 0][idx] = (_Float16)v.x;
      sWh[tile][lb + 1][idx] = (_Float16)v.y;
      sWh[tile][lb + 2][idx] = (_Float16)v.z;
      sWh[tile][lb + 3][idx] = (_Float16)v.w;
    }
    __syncthreads();

    // prefetch next K chunk of A (global_prefetch_b8)
    if (kc + 32 < K)
      __builtin_prefetch(A + (size_t)(m0 + (threadIdx.x >> 1)) * K + kc + 32, 0, 1);

    v16h af = *(const v16h*)(&sAh[wave][lane][0]);
#pragma unroll
    for (int nt = 0; nt < 4; ++nt) {
      v16h bf = *(const v16h*)(&sWh[nt][lane][0]);
      acc[nt] = wmma_f16(af, bf, acc[nt]);
    }
    __syncthreads();
  }

  int colb   = lane & 15;
  int rowoff = (lane >> 4) * 8;
#pragma unroll
  for (int nt = 0; nt < 4; ++nt) {
#pragma unroll
    for (int r = 0; r < 8; ++r) {
      int row = m0 + wave * 16 + rowoff + r;
      int col = n0 + nt * 16 + colb;
      float v = acc[nt][r];
      if (bias) v += bias[col];
      if (act == 1)      v = fmaxf(v, 0.f);
      else if (act == 2) v = 1.f / (1.f + __expf(-v));
      C[(size_t)row * N + col] = v;
    }
  }
}

// ---------------------------------------------------------------------------
// 2) Attention: one wave handles one (b, h, 16-row q tile).
// ---------------------------------------------------------------------------
__global__ __launch_bounds__(32)
void ngfkt_attn_kernel(const float* __restrict__ qb, const float* __restrict__ kb,
                       const float* __restrict__ vb,
                       const float* __restrict__ cg, const float* __restrict__ gg,
                       const float* __restrict__ rel, const float* __restrict__ ts,
                       const float* __restrict__ posK, const float* __restrict__ posV,
                       const float* __restrict__ l1p, const float* __restrict__ l2p,
                       float* __restrict__ outb, float* __restrict__ attn_out,
                       int layer) {
  __shared__ float sS[16][SS];       // scores -> probs
  __shared__ float sStage[32][32];   // K/V staging (async-LDS)
  __shared__ float sTab[16][10];     // q . pos_key_emb[p]
  __shared__ float sPv[16][10];      // prob mass per relative-pos bucket

  int lane = threadIdx.x & 31;
  int tile = blockIdx.x;             // B*H*(S/16)
  int qt = tile & 31;
  int h  = (tile >> 5) & 7;
  int b  = tile >> 8;
  int q0 = qt * 16;

  const float l1 = *l1p, l2 = *l2p;
  const float w_sm  = (1.f - l2) * (1.f - l1);
  const float w_rel = (1.f - l2) * l1;
  const float w_t   = l2;
  const float scale = 0.17677669529663687f;  // 1/sqrt(32)

  // ---- q fragment (16x32 f16) straight from global
  v16h afrag;
  {
    int m  = lane & 15;
    int kh = (lane >> 4) * 8;
    const float* qr = qb + ((size_t)(b * SS + q0 + m)) * EE + h * DD;
#pragma unroll
    for (int i = 0; i < 8; ++i) {
      afrag[i]     = (_Float16)qr[kh + i];
      afrag[i + 8] = (_Float16)qr[16 + kh + i];
    }
  }

  // ---- per-row posK dot table + bucket init
  if (lane < 16) {
    const float* qr = qb + ((size_t)(b * SS + q0 + lane)) * EE + h * DD;
    for (int p = 0; p < 10; ++p) {
      float s = 0.f;
      for (int d = 0; d < DD; ++d) s += qr[d] * posK[p * DD + d];
      sTab[lane][p] = s;
      sPv[lane][p]  = 0.f;
    }
  }
  __syncthreads();

  // ---- scores: 32 key tiles of 16, K=D=32 -> one WMMA each
  for (int kt = 0; kt < 32; ++kt) {
    int k0 = kt * 16;
    async_stage_f32(kb + ((size_t)(b * SS + k0)) * EE + h * DD, 16, 32, EE,
                    &sStage[0][0], 32);
    __syncthreads();
    v16h bfrag = load_b_frag_T(&sStage[0][0], 32);
    v8f sc = {};
    sc = wmma_f16(afrag, bfrag, sc);

    int n = lane & 15;
    int rowoff = (lane >> 4) * 8;
    int kabs = k0 + n;
#pragma unroll
    for (int r = 0; r < 8; ++r) {
      int row  = rowoff + r;
      int qabs = q0 + row;
      int p = qabs - kabs; p = p < 0 ? 0 : (p > 9 ? 9 : p);
      float val = (sc[r] + sTab[row][p]) * scale;
      if (kabs > qabs) val = NEGV;
      sS[row][kabs] = val;
    }
    __syncthreads();
  }

  // ---- softmax + mixing with rel_attn and time_attn (per row, lanes 0..15)
  if (lane < 16) {
    int qabs = q0 + lane;
    float* row = &sS[lane][0];
    float mx = -1e30f;
    for (int k2 = 0; k2 < SS; ++k2) mx = fmaxf(mx, row[k2]);
    float sum = 0.f;
    for (int k2 = 0; k2 < SS; ++k2) { float e = __expf(row[k2] - mx); row[k2] = e; sum += e; }
    float inv = 1.f / sum;

    const float* rrow = rel + ((size_t)b * SS + qabs) * SS;
    const float* trow = ts  + ((size_t)b * SS + qabs) * SS;

    float mx2 = -1e30f;
    for (int k2 = 0; k2 < SS; ++k2) {
      float rm = (k2 <= qabs) ? rrow[k2] : 0.f;
      float vv = (rm == 0.f) ? -1e5f : rm;
      mx2 = fmaxf(mx2, vv);
    }
    float sum2 = 0.f;
    for (int k2 = 0; k2 < SS; ++k2) {
      float rm = (k2 <= qabs) ? rrow[k2] : 0.f;
      float vv = (rm == 0.f) ? -1e5f : rm;
      sum2 += __expf(vv - mx2);
    }
    float inv2 = 1.f / sum2;

    float mx3 = -1e30f, sum3 = 0.f, inv3 = 0.f, invq = 0.f;
    if (layer == 0) {
      for (int k2 = 0; k2 < SS; ++k2) {
        float tv = (k2 <= qabs) ? __expf(-fabsf(trow[k2])) : -1e5f;
        mx3 = fmaxf(mx3, tv);
      }
      for (int k2 = 0; k2 < SS; ++k2) {
        float tv = (k2 <= qabs) ? __expf(-fabsf(trow[k2])) : -1e5f;
        sum3 += __expf(tv - mx3);
      }
      inv3 = 1.f / sum3;
    } else {
      invq = 1.f / (float)(qabs + 1);  // exp(0) masked softmax -> uniform
    }

    for (int k2 = 0; k2 < SS; ++k2) {
      float sm = row[k2] * inv;
      float rm = (k2 <= qabs) ? rrow[k2] : 0.f;
      float rv = __expf(((rm == 0.f) ? -1e5f : rm) - mx2) * inv2;
      float tvp;
      if (layer == 0) {
        float t0 = (k2 <= qabs) ? __expf(-fabsf(trow[k2])) : -1e5f;
        tvp = __expf(t0 - mx3) * inv3;
      } else {
        tvp = (k2 <= qabs) ? invq : 0.f;
      }
      row[k2] = w_sm * sm + w_rel * rv + w_t * tvp;
    }
  }
  __syncthreads();

  // ---- export attention probabilities (last layer only)
  if (attn_out) {
    float* dst = attn_out + (((size_t)b * HH + h) * SS + q0) * SS;
    const float* src = &sS[0][0];
    for (int t = lane; t < 16 * SS; t += 32) dst[t] = src[t];
  }

  // ---- relative-position bucket masses for posV term
  if (lane < 16) {
    int qabs = q0 + lane;
    for (int k2 = 0; k2 <= qabs; ++k2) {
      int p = qabs - k2; if (p > 9) p = 9;
      sPv[lane][p] += sS[lane][k2];
    }
  }
  __syncthreads();

  // ---- out = prob @ v   (16 K-chunks of 32, two 16-col D tiles)
  v8f acc0 = {}, acc1 = {};
  for (int kt = 0; kt < 16; ++kt) {
    int k0 = kt * 32;
    async_stage_f32(vb + ((size_t)(b * SS + k0)) * EE + h * DD, 32, 32, EE,
                    &sStage[0][0], 32);
    __syncthreads();
    v16h ap = load_a_frag(&sS[0][k0], SS);
    v16h b0 = load_b_frag(&sStage[0][0], 32, 0);
    v16h b1 = load_b_frag(&sStage[0][0], 32, 16);
    acc0 = wmma_f16(ap, b0, acc0);
    acc1 = wmma_f16(ap, b1, acc1);
    __syncthreads();
  }

  // ---- epilogue: posV term + guess/slip gating
  {
    int rowoff = (lane >> 4) * 8;
    int col = lane & 15;
#pragma unroll
    for (int r = 0; r < 8; ++r) {
      int row  = rowoff + r;
      int qabs = q0 + row;
      size_t base = ((size_t)b * SS + qabs) * EE + h * DD;
      float pv0 = 0.f, pv1 = 0.f;
      for (int p = 0; p < 10; ++p) {
        float s = sPv[row][p];
        pv0 += s * posV[p * DD + col];
        pv1 += s * posV[p * DD + 16 + col];
      }
      float v0 = acc0[r] + pv0;
      float v1 = acc1[r] + pv1;
      float c0 = cg[base + col],      c1 = cg[base + 16 + col];
      float g0 = gg[base + col],      g1 = gg[base + 16 + col];
      outb[base + col]      = g0 + (1.f - c0) * v0;
      outb[base + 16 + col] = g1 + (1.f - c1) * v1;
    }
  }
}

// ---------------------------------------------------------------------------
// 3) outputs += relu(residual)
// ---------------------------------------------------------------------------
__global__ __launch_bounds__(256)
void ngfkt_addrelu_kernel(float* __restrict__ a, const float* __restrict__ r, int n) {
  int i = blockIdx.x * blockDim.x + threadIdx.x;
  if (i < n) a[i] += fmaxf(r[i], 0.f);
}

// ---------------------------------------------------------------------------
// 4) final: out[row] = dot(outputs[row,:], lin_out_w) + b   (one wave per row)
// ---------------------------------------------------------------------------
__global__ __launch_bounds__(256)
void ngfkt_final_kernel(const float* __restrict__ x, const float* __restrict__ w,
                        const float* __restrict__ bias, float* __restrict__ out) {
  int wave = threadIdx.x >> 5;
  int lane = threadIdx.x & 31;
  int row  = blockIdx.x * 8 + wave;
  const float* xr = x + (size_t)row * EE;
  float s = 0.f;
  for (int i = lane; i < EE; i += 32) s += xr[i] * w[i];
#pragma unroll
  for (int off = 16; off > 0; off >>= 1) s += __shfl_xor(s, off);
  if (lane == 0) out[row] = s + bias[0];
}

// ---------------------------------------------------------------------------
extern "C" void kernel_launch(void* const* d_in, const int* in_sizes, int n_in,
                              void* d_out, int out_size, void* d_ws, size_t ws_size,
                              hipStream_t stream) {
  (void)in_sizes; (void)n_in; (void)out_size; (void)ws_size;

  const int*   item_inputs  = (const int*)d_in[2];
  const int*   label_inputs = (const int*)d_in[3];
  const int*   item_ids     = (const int*)d_in[4];
  const float* rel      = (const float*)d_in[5];
  const float* ts       = (const float*)d_in[6];
  const float* item_emb = (const float*)d_in[7];
  const float* posK     = (const float*)d_in[8];
  const float* posV     = (const float*)d_in[9];
  const float* Wc       = (const float*)d_in[10];
  const float* Wg       = (const float*)d_in[11];
  const float* lin_in_w = (const float*)d_in[12];
  const float* lin_in_b = (const float*)d_in[13];
  const float* Wq       = (const float*)d_in[14];
  const float* bq       = (const float*)d_in[15];
  const float* Wk       = (const float*)d_in[16];
  const float* bk       = (const float*)d_in[17];
  const float* Wv       = (const float*)d_in[18];
  const float* bv       = (const float*)d_in[19];
  const float* l1       = (const float*)d_in[20];
  const float* l2       = (const float*)d_in[21];
  const float* lout_w   = (const float*)d_in[22];
  const float* lout_b   = (const float*)d_in[23];

  const int M = BB * SS;      // 8192
  const int NE = M * EE;      // 2,097,152

  float* ws      = (float*)d_ws;
  float* inputs  = ws;            ws += (size_t)M * 2 * EE;
  float* query   = ws;            ws += NE;
  float* xbuf    = ws;            ws += NE;
  float* cgate   = ws;            ws += NE;
  float* ggate   = ws;            ws += NE;
  float* qbuf    = ws;            ws += NE;
  float* kbuf    = ws;            ws += NE;
  float* vbuf    = ws;            ws += NE;
  float* out0    = ws;            ws += NE;
  float* out1    = ws;            ws += NE;

  float* final_out = (float*)d_out;
  float* attn_out  = (float*)d_out + M;   // [B,H,S,S]

  dim3 gG((M * EE + 255) / 256);
  ngfkt_gather_kernel<<<gG, 256, 0, stream>>>(item_emb, item_inputs, label_inputs,
                                              item_ids, inputs, query);

  dim3 gemmGrid(M / 64, EE / 64);  // (128, 4)
  // x = relu(inputs @ lin_in_w + b)
  ngfkt_gemm_kernel<<<gemmGrid, 128, 0, stream>>>(inputs, lin_in_w, lin_in_b, xbuf,
                                                  M, EE, 2 * EE, 1);
  // gates
  ngfkt_gemm_kernel<<<gemmGrid, 128, 0, stream>>>(query, Wc, nullptr, cgate, M, EE, EE, 2);
  ngfkt_gemm_kernel<<<gemmGrid, 128, 0, stream>>>(query, Wg, nullptr, ggate, M, EE, EE, 2);

  const int attnBlocks = BB * HH * (SS / 16);  // 4096

  // ---- layer 0 (kv = x, real timestamps)
  ngfkt_gemm_kernel<<<gemmGrid, 128, 0, stream>>>(query, Wq, bq, qbuf, M, EE, EE, 0);
  ngfkt_gemm_kernel<<<gemmGrid, 128, 0, stream>>>(xbuf,  Wk, bk, kbuf, M, EE, EE, 0);
  ngfkt_gemm_kernel<<<gemmGrid, 128, 0, stream>>>(xbuf,  Wv, bv, vbuf, M, EE, EE, 0);
  ngfkt_attn_kernel<<<attnBlocks, 32, 0, stream>>>(qbuf, kbuf, vbuf, cgate, ggate,
                                                   rel, ts, posK, posV, l1, l2,
                                                   out0, nullptr, 0);

  // ---- layer 1 (kv = out0, timestamps = 0 -> uniform time attn); exports attn
  const size_t wOff = (size_t)EE * EE;
  ngfkt_gemm_kernel<<<gemmGrid, 128, 0, stream>>>(query, Wq + wOff, bq + EE, qbuf, M, EE, EE, 0);
  ngfkt_gemm_kernel<<<gemmGrid, 128, 0, stream>>>(out0,  Wk + wOff, bk + EE, kbuf, M, EE, EE, 0);
  ngfkt_gemm_kernel<<<gemmGrid, 128, 0, stream>>>(out0,  Wv + wOff, bv + EE, vbuf, M, EE, EE, 0);
  ngfkt_attn_kernel<<<attnBlocks, 32, 0, stream>>>(qbuf, kbuf, vbuf, cgate, ggate,
                                                   rel, ts, posK, posV, l1, l2,
                                                   out1, attn_out, 1);

  // outputs = out0 + relu(out1)
  ngfkt_addrelu_kernel<<<(NE + 255) / 256, 256, 0, stream>>>(out0, out1, NE);

  // final projection
  ngfkt_final_kernel<<<M / 8, 256, 0, stream>>>(out0, lout_w, lout_b, final_out);
}
